// WGAT_14508399525893
// MI455X (gfx1250) — matrix operand order repaired and checked by reading, not compiled, not run on hardware
//
#include <hip/hip_runtime.h>
#include <hip/hip_bf16.h>
#include <math.h>

typedef __attribute__((ext_vector_type(16))) __bf16 v16bf;
typedef __attribute__((ext_vector_type(8)))  __bf16 v8bf;
typedef __attribute__((ext_vector_type(8)))  float  v8f;
typedef int v4i __attribute__((vector_size(4 * sizeof(int))));

#define BN    2
#define CCH   128
#define HH    128
#define WWID  128
#define WS8   8
#define N1    16
#define NNODE 256               // nodes per batch
#define NODES (BN*NNODE)        // 512
#define PATCH (CCH*WS8*WS8)     // 8192
#define KNN   8

#if defined(__has_builtin)
#if __has_builtin(__builtin_amdgcn_global_load_async_to_lds_b128)
#define HAVE_ASYNC_LDS 1
#endif
#endif

__device__ __forceinline__ float gelu_f(float v) {
    return 0.5f * v * (1.0f + erff(v * 0.70710678118654752f));
}

// ---------------------------------------------------------------------------
// 1) image -> node patches: f32 patch-major s/d + bf16 channels-last mirrors
//    + |feat|^2 per node
// ---------------------------------------------------------------------------
__global__ __launch_bounds__(256) void k_pack(const float* __restrict__ x,
                                              float* __restrict__ sbuf,
                                              float* __restrict__ dbuf,
                                              __bf16* __restrict__ sbf,
                                              __bf16* __restrict__ dbf,
                                              float* __restrict__ sq) {
    int node = blockIdx.x, tid = threadIdx.x;
    int bi = node >> 8, nl = node & 255, py = nl >> 4, px = nl & 15;
    float acc = 0.f;
    for (int i = 0; i < 32; ++i) {
        int e = i * 256 + tid;
        int ci = e >> 6, p = e & 63, iy = p >> 3, ix = p & 7;
        float v = x[(((size_t)bi * CCH + ci) * HH + py * WS8 + iy) * WWID + px * WS8 + ix];
        size_t o = (size_t)node * PATCH;
        sbuf[o + e] = v; dbuf[o + e] = v;
        __bf16 bv = (__bf16)v;
        sbf[o + p * CCH + ci] = bv;          // channels-last
        dbf[o + p * CCH + ci] = bv;
        acc += v * v;
    }
    __shared__ float red[256];
    red[tid] = acc; __syncthreads();
    for (int s = 128; s > 0; s >>= 1) { if (tid < s) red[tid] += red[tid + s]; __syncthreads(); }
    if (tid == 0) sq[node] = red[0];
}

// ---------------------------------------------------------------------------
// 2) cdist via bf16 WMMA GEMM on channels-last features (K-permutation
//    invariant): G = F*F^T ; D = sqrt(max(si+sj-2G,1e-12))
// ---------------------------------------------------------------------------
__global__ __launch_bounds__(256) void k_cdist(const __bf16* __restrict__ featbf,
                                               const float* __restrict__ sq,
                                               float* __restrict__ Dm) {
    int wave = blockIdx.x * 8 + (threadIdx.x >> 5);
    int lane = threadIdx.x & 31;
    int bi = wave >> 8;
    int tl = wave & 255, ti = tl >> 4, tj = tl & 15;
    int m = lane & 15, hi = lane >> 4;
    const __bf16* rowA = featbf + (size_t)(bi * NNODE + ti * 16 + m) * PATCH;
    const __bf16* rowB = featbf + (size_t)(bi * NNODE + tj * 16 + m) * PATCH;
    v8f acc = {};
    for (int kc = 0; kc < PATCH / 32; ++kc) {
        int ka = kc * 32 + hi * 8;
        v8bf a0 = *(const v8bf*)(rowA + ka);
        v8bf a1 = *(const v8bf*)(rowA + ka + 16);
        int kb = kc * 32 + hi * 8;
        v8bf b0 = *(const v8bf*)(rowB + kb);
        v8bf b1 = *(const v8bf*)(rowB + kb + 16);
        v16bf A, B;
#pragma unroll
        for (int j = 0; j < 8; ++j) { A[j] = a0[j]; A[j + 8] = a1[j]; B[j] = b0[j]; B[j + 8] = b1[j]; }
        acc = __builtin_amdgcn_wmma_f32_16x16x32_bf16(false, A, false, B, (short)0, acc, false, false);
    }
#pragma unroll
    for (int v = 0; v < 8; ++v) {
        int i = ti * 16 + v + 8 * hi;
        int j = tj * 16 + m;
        float d2 = sq[bi * NNODE + i] + sq[bi * NNODE + j] - 2.f * acc[v];
        Dm[((size_t)bi * NNODE + i) * NNODE + j] = sqrtf(fmaxf(d2, 1e-12f));
    }
}

// ---------------------------------------------------------------------------
// 3) kNN (k=8 smallest incl. self, low-index ties) + edge weights e
// ---------------------------------------------------------------------------
__global__ __launch_bounds__(256) void k_knn(const float* __restrict__ Dm,
                                             int* __restrict__ idx,
                                             float* __restrict__ ebuf) {
    int t = blockIdx.x * 256 + threadIdx.x;    // 512 rows
    int bi = t >> 8, i = t & 255;
    const float* row = Dm + ((size_t)bi * NNODE + i) * NNODE;
    for (int j = 0; j < NNODE; j += 32) __builtin_prefetch(row + j, 0, 3);
    unsigned sel[8] = {0, 0, 0, 0, 0, 0, 0, 0};
    float dist[KNN];
    for (int kk = 0; kk < KNN; ++kk) {
        float best = 3.4e38f; int bj = 0;
        for (int j = 0; j < NNODE; ++j) {
            if ((sel[j >> 5] >> (j & 31)) & 1u) continue;
            float dv = row[j];
            if (dv < best) { best = dv; bj = j; }
        }
        sel[bj >> 5] |= 1u << (bj & 31);
        idx[t * KNN + kk] = bj; dist[kk] = best;
    }
    float sg = 0.f;
    for (int kk = 0; kk < KNN; ++kk) sg += dist[kk];
    sg *= (1.f / KNN);
    float inv = 1.f / (sg * sg);
    for (int kk = 0; kk < KNN; ++kk) ebuf[t * KNN + kk] = expf(-dist[kk] * inv);
}

// ---------------------------------------------------------------------------
// weight repack: OIHW f32 -> bf16 [co][(ky*3+kx)*Ci + ci]   (Co = 128)
// ---------------------------------------------------------------------------
__global__ void k_wconv(const float* __restrict__ w, __bf16* __restrict__ out, int Ci) {
    int t = blockIdx.x * blockDim.x + threadIdx.x;
    int per = Ci * 9;
    if (t >= CCH * per) return;
    int co = t / per, r = t - co * per;
    int ci = r / 9, q = r - ci * 9;                // q = ky*3+kx
    out[co * per + q * Ci + ci] = (__bf16)w[t];
}

// ---------------------------------------------------------------------------
// 4) per-node 3x3 conv (pad 1) as implicit GEMM with bf16 WMMA.
//    Activations staged into a zero-padded 10x10x128 channels-last LDS tile
//    (interior via CDNA5 async global->LDS, halo zeroed via DS — disjoint
//    addresses, so no ordering hazard). Inner loop is mask-free.
// ---------------------------------------------------------------------------
__global__ __launch_bounds__(256) void k_conv3x3(const __bf16* __restrict__ inbf,
                                                 const __bf16* __restrict__ Wb,
                                                 const float* __restrict__ bias,
                                                 float* __restrict__ out, int mode) {
    __shared__ __bf16 lds[10 * 10 * CCH];          // padded [y+1][x+1][ci] : 25 KB
    int node = blockIdx.x, tid = threadIdx.x;
    const __bf16* gsrc = inbf + (size_t)node * PATCH;
    // zero the halo: 36 border cells x 128 ch = 4608 elements
    for (int i = 0; i < 18; ++i) {
        int t = i * 256 + tid;
        int cell = t >> 7, ch = t & 127;
        int yy, xx;
        if (cell < 10)      { yy = 0; xx = cell; }
        else if (cell < 20) { yy = 9; xx = cell - 10; }
        else { int r = cell - 20; yy = 1 + (r >> 1); xx = (r & 1) * 9; }
        lds[(yy * 10 + xx) * CCH + ch] = (__bf16)0.f;
    }
#ifdef HAVE_ASYNC_LDS
#pragma unroll
    for (int i = 0; i < 4; ++i) {                  // 1024 x 16B = 16 KB interior
        int t2 = i * 256 + tid;
        int row = t2 >> 7, c16 = t2 & 127;         // row 0..7, 128 x 16B per row
        __builtin_amdgcn_global_load_async_to_lds_b128(
            (__attribute__((address_space(1))) v4i*)(const_cast<__bf16*>(gsrc) + row * 1024 + c16 * 8),
            (__attribute__((address_space(3))) v4i*)(lds + ((row + 1) * 10 + 1) * CCH + c16 * 8),
            0, 0);
    }
    asm volatile("s_wait_asynccnt 0x0" ::: "memory");
#else
    for (int i = 0; i < 32; ++i) {
        int t = i * 256 + tid;                     // t = p*128 + ci
        int ci = t & 127, p = t >> 7, yy = p >> 3, xx = p & 7;
        lds[((yy + 1) * 10 + (xx + 1)) * CCH + ci] = gsrc[t];
    }
#endif
    __syncthreads();

    int lane = tid & 31, cot = tid >> 5;
    int m = lane & 15, hi = lane >> 4;
    const __bf16* wrow = Wb + (size_t)(cot * 16 + m) * (CCH * 9);
    __builtin_prefetch(wrow, 0, 3);
    v8f acc0 = {}, acc1 = {}, acc2 = {}, acc3 = {};
    for (int kc = 0; kc < 36; ++kc) {              // K = 1152
        int ka = kc * 32 + hi * 8;
        v8bf a0 = *(const v8bf*)(wrow + ka);
        v8bf a1 = *(const v8bf*)(wrow + ka + 16);
        v16bf A;
#pragma unroll
        for (int j = 0; j < 8; ++j) { A[j] = a0[j]; A[j + 8] = a1[j]; }
        int f = kc >> 2;                           // (ky,kx) fixed per chunk
        int ky = f / 3, kx = f - ky * 3;
        int cib = (kc & 3) * 32 + hi * 8;          // ci runs cib..cib+7 and cib+16..cib+23
#pragma unroll
        for (int nt = 0; nt < 4; ++nt) {
            int p = nt * 16 + m;
            int oy = p >> 3, ox = p & 7;
            int base = ((oy + ky) * 10 + (ox + kx)) * CCH + cib;   // always in-bounds
            v8bf b0 = *(const v8bf*)(&lds[base]);
            v8bf b1 = *(const v8bf*)(&lds[base + 16]);
            v16bf B;
#pragma unroll
            for (int j = 0; j < 8; ++j) { B[j] = b0[j]; B[j + 8] = b1[j]; }
            v8f& acc = nt == 0 ? acc0 : nt == 1 ? acc1 : nt == 2 ? acc2 : acc3;
            acc = __builtin_amdgcn_wmma_f32_16x16x32_bf16(false, A, false, B, (short)0, acc, false, false);
        }
    }
    float* dst = out + (size_t)node * PATCH;
#pragma unroll
    for (int nt = 0; nt < 4; ++nt) {
        v8f acc = nt == 0 ? acc0 : nt == 1 ? acc1 : nt == 2 ? acc2 : acc3;
#pragma unroll
        for (int v = 0; v < 8; ++v) {
            int co = cot * 16 + v + 8 * hi;
            int p = nt * 16 + m;
            float r = acc[v];
            if (mode == 1) r = gelu_f(r + bias[co]);
            dst[co * 64 + p] = r;
        }
    }
}

// ---------------------------------------------------------------------------
// 5) attention scalars: a_src/a_dst = (1/16) * weighted sum of d over patch
// ---------------------------------------------------------------------------
__global__ __launch_bounds__(256) void k_atten(const float* __restrict__ dcur,
                                               const float* __restrict__ aw,
                                               float* __restrict__ asrc,
                                               float* __restrict__ adst) {
    int node = blockIdx.x, tid = threadIdx.x;
    const float* src = dcur + (size_t)node * PATCH;
    float s0 = 0.f, s1 = 0.f;
    for (int i = 0; i < 32; ++i) {
        int e = i * 256 + tid;
        int ci = e >> 6, p = e & 63, y = p >> 3, x = p & 7;
        int wo = ci * 4 + (y & 1) * 2 + (x & 1);
        float v = src[e];
        s0 += aw[wo] * v;
        s1 += aw[CCH * 4 + wo] * v;
    }
    __shared__ float r0[256], r1[256];
    r0[tid] = s0; r1[tid] = s1; __syncthreads();
    for (int s = 128; s > 0; s >>= 1) {
        if (tid < s) { r0[tid] += r0[tid + s]; r1[tid] += r1[tid + s]; }
        __syncthreads();
    }
    if (tid == 0) { asrc[node] = r0[0] * (1.f / 16.f); adst[node] = r1[0] * (1.f / 16.f); }
}

// ---------------------------------------------------------------------------
// 6) leaky-relu + softmax over k=8 mailbox
// ---------------------------------------------------------------------------
__global__ void k_softmax(const float* __restrict__ asrc, const float* __restrict__ adst,
                          const int* __restrict__ idx, const float* __restrict__ bias,
                          float* __restrict__ alpha) {
    int t = blockIdx.x * blockDim.x + threadIdx.x;
    if (t >= NODES) return;
    int bi = t >> 8;
    float b = bias[0], ad = adst[t];
    float a[KNN], mx = -3.4e38f;
    for (int kk = 0; kk < KNN; ++kk) {
        int j = idx[t * KNN + kk];
        float v = asrc[bi * NNODE + j] + ad + b;
        v = v < 0.f ? 0.01f * v : v;
        a[kk] = v; mx = fmaxf(mx, v);
    }
    float s = 0.f;
    for (int kk = 0; kk < KNN; ++kk) { a[kk] = expf(a[kk] - mx); s += a[kk]; }
    float inv = 1.f / s;
    for (int kk = 0; kk < KNN; ++kk) alpha[t * KNN + kk] = a[kk] * inv;
}

// ---------------------------------------------------------------------------
// 7) aggregation: s = max_k GELU(e*ms[nb]+b) ; d = sum_k alpha*d[nb]
//    writes f32 + bf16 channels-last mirrors (next layer's WMMA inputs)
// ---------------------------------------------------------------------------
__global__ __launch_bounds__(256) void k_aggregate(const float* __restrict__ dnew,
                                                   const float* __restrict__ ms,
                                                   const float* __restrict__ ebuf,
                                                   const float* __restrict__ alpha,
                                                   const int* __restrict__ idx,
                                                   const float* __restrict__ sbias,
                                                   float* __restrict__ sout,
                                                   float* __restrict__ dout_,
                                                   __bf16* __restrict__ sbf,
                                                   __bf16* __restrict__ dbf) {
    int node = blockIdx.x, tid = threadIdx.x;
    int bi = node >> 8;
    int nb[KNN]; float ec[KNN], al[KNN];
    for (int kk = 0; kk < KNN; ++kk) {
        nb[kk] = bi * NNODE + idx[node * KNN + kk];
        ec[kk] = ebuf[node * KNN + kk];
        al[kk] = alpha[node * KNN + kk];
    }
    for (int i = 0; i < 32; ++i) {
        int e = i * 256 + tid; int ci = e >> 6, p = e & 63;
        float bsc = sbias[ci];
        float smax = -3.4e38f, dsum = 0.f;
        for (int kk = 0; kk < KNN; ++kk) {
            size_t go = (size_t)nb[kk] * PATCH + e;
            float mv = gelu_f(ec[kk] * ms[go] + bsc);
            smax = fmaxf(smax, mv);
            dsum += al[kk] * dnew[go];
        }
        size_t o = (size_t)node * PATCH;
        sout[o + e] = smax;
        dout_[o + e] = dsum;
        sbf[o + p * CCH + ci] = (__bf16)smax;
        dbf[o + p * CCH + ci] = (__bf16)dsum;
    }
}

// ---------------------------------------------------------------------------
// 8) final 3x3 conv (cat[s,d]: 256 -> 128 ch) over full image, bf16 WMMA.
//    Reads the channels-last node buffers directly (no image materialization).
// ---------------------------------------------------------------------------
__global__ __launch_bounds__(256) void k_outconv(const __bf16* __restrict__ sbf,
                                                 const __bf16* __restrict__ dbf,
                                                 const __bf16* __restrict__ Wb,
                                                 const float* __restrict__ bias,
                                                 float* __restrict__ out) {
    int bid = blockIdx.x;
    int xt = bid & 7, y = (bid >> 3) & 127, bi = bid >> 10;
    int tid = threadIdx.x, lane = tid & 31, cot = tid >> 5;
    int m = lane & 15, hi = lane >> 4;
    const __bf16* wrow = Wb + (size_t)(cot * 16 + m) * (2 * CCH * 9);
    __builtin_prefetch(wrow, 0, 3);
    v8f acc = {};
    for (int kc = 0; kc < 72; ++kc) {              // K = 2304
        int ka = kc * 32 + hi * 8;
        v8bf a0 = *(const v8bf*)(wrow + ka);
        v8bf a1 = *(const v8bf*)(wrow + ka + 16);
        v16bf A;
#pragma unroll
        for (int j = 0; j < 8; ++j) { A[j] = a0[j]; A[j + 8] = a1[j]; }
        int f = kc >> 3;                           // (ky,kx) fixed per chunk
        int ky = f / 3, kx = f - ky * 3;
        int cib = (kc & 7) * 32 + hi * 8;          // 0..232, 8-aligned
        int yy = y + ky - 1;
        int xx = xt * 16 + m + kx - 1;
        bool inb = ((unsigned)yy < 128u) && ((unsigned)xx < 128u);
        int yyc = inb ? yy : 0, xxc = inb ? xx : 0;
        int node = bi * NNODE + (yyc >> 3) * N1 + (xxc >> 3);
        int p = (yyc & 7) * 8 + (xxc & 7);
        const __bf16* srcb = (cib < CCH) ? sbf : dbf;
        const __bf16* bp = srcb + (size_t)node * PATCH + p * CCH + (cib & 127);
        v8bf b0 = *(const v8bf*)bp;
        v8bf b1 = *(const v8bf*)(bp + 16);
        v16bf B;
#pragma unroll
        for (int j = 0; j < 8; ++j) {
            B[j]     = inb ? b0[j] : (__bf16)0.f;
            B[j + 8] = inb ? b1[j] : (__bf16)0.f;
        }
        acc = __builtin_amdgcn_wmma_f32_16x16x32_bf16(false, A, false, B, (short)0, acc, false, false);
    }
#pragma unroll
    for (int v = 0; v < 8; ++v) {
        int co = cot * 16 + v + 8 * hi;
        int x = xt * 16 + m;
        out[(((size_t)bi * CCH + co) * HH + y) * WWID + x] = gelu_f(acc[v] + bias[co]);
    }
}

// ---------------------------------------------------------------------------
extern "C" void kernel_launch(void* const* d_in, const int* in_sizes, int n_in,
                              void* d_out, int out_size, void* d_ws, size_t ws_size,
                              hipStream_t stream) {
    const float* x   = (const float*)d_in[0];
    const float* csw = (const float*)d_in[1];
    const float* csb = (const float*)d_in[2];
    const float* cdw = (const float*)d_in[3];
    const float* cdb = (const float*)d_in[4];
    const float* aw  = (const float*)d_in[5];
    const float* ab  = (const float*)d_in[6];
    const float* ow  = (const float*)d_in[7];
    const float* ob  = (const float*)d_in[8];
    float* outp = (float*)d_out;

    // workspace layout (~86 MB)
    char* ws = (char*)d_ws;
    size_t off = 0;
    auto alloc = [&](size_t bytes) -> char* {
        char* p = ws + off; off = (off + bytes + 255) & ~(size_t)255; return p;
    };
    float*  sbuf  = (float*) alloc((size_t)NODES * PATCH * 4);
    float*  dbuf  = (float*) alloc((size_t)NODES * PATCH * 4);
    float*  tmpA  = (float*) alloc((size_t)NODES * PATCH * 4);   // conv_d out (new d)
    float*  tmpB  = (float*) alloc((size_t)NODES * PATCH * 4);   // ms
    __bf16* sbf   = (__bf16*)alloc((size_t)NODES * PATCH * 2);   // channels-last bf16 s
    __bf16* dbf   = (__bf16*)alloc((size_t)NODES * PATCH * 2);   // channels-last bf16 d
    float*  Dm    = (float*) alloc((size_t)BN * NNODE * NNODE * 4);
    float*  sq    = (float*) alloc((size_t)NODES * 4);
    int*    idxb  = (int*)   alloc((size_t)NODES * KNN * 4);
    float*  ebuf  = (float*) alloc((size_t)NODES * KNN * 4);
    float*  albuf = (float*) alloc((size_t)NODES * KNN * 4);
    float*  asrc  = (float*) alloc((size_t)NODES * 4);
    float*  adst  = (float*) alloc((size_t)NODES * 4);
    __bf16* WbS[2]; __bf16* WbD[2];
    for (int l = 0; l < 2; ++l) {
        WbS[l] = (__bf16*)alloc((size_t)CCH * CCH * 9 * 2);
        WbD[l] = (__bf16*)alloc((size_t)CCH * CCH * 9 * 2);
    }
    __bf16* WbO = (__bf16*)alloc((size_t)CCH * 2 * CCH * 9 * 2);
    (void)ws_size; (void)in_sizes; (void)n_in; (void)out_size;

    // weight repacks (f32 OIHW -> bf16 [co][(ky*3+kx)*Ci+ci])
    for (int l = 0; l < 2; ++l) {
        k_wconv<<<(CCH * CCH * 9 + 255) / 256, 256, 0, stream>>>(csw + (size_t)l * CCH * CCH * 9, WbS[l], CCH);
        k_wconv<<<(CCH * CCH * 9 + 255) / 256, 256, 0, stream>>>(cdw + (size_t)l * CCH * CCH * 9, WbD[l], CCH);
    }
    k_wconv<<<(CCH * 2 * CCH * 9 + 255) / 256, 256, 0, stream>>>(ow, WbO, 2 * CCH);

    // graph construction
    k_pack<<<NODES, 256, 0, stream>>>(x, sbuf, dbuf, sbf, dbf, sq);
    k_cdist<<<(BN * 256) / 8, 256, 0, stream>>>(sbf, sq, Dm);
    k_knn<<<NODES / 256, 256, 0, stream>>>(Dm, idxb, ebuf);

    // 2 GNN layers
    for (int l = 0; l < 2; ++l) {
        k_conv3x3<<<NODES, 256, 0, stream>>>(dbf, WbD[l], cdb + (size_t)l * CCH, tmpA, 1);
        k_atten<<<NODES, 256, 0, stream>>>(tmpA, aw + (size_t)l * 2 * CCH * 4, asrc, adst);
        k_softmax<<<NODES / 256, 256, 0, stream>>>(asrc, adst, idxb, ab + l, albuf);
        k_conv3x3<<<NODES, 256, 0, stream>>>(sbf, WbS[l], nullptr, tmpB, 0);
        k_aggregate<<<NODES, 256, 0, stream>>>(tmpA, tmpB, ebuf, albuf, idxb,
                                               csb + (size_t)l * CCH, sbuf, dbuf, sbf, dbf);
    }

    // output conv straight from channels-last node buffers
    k_outconv<<<BN * HH * 8, 256, 0, stream>>>(sbf, dbf, WbO, ob, outp);
}
